// CRF_66623532695626
// MI455X (gfx1250) — compile-verified, matrix operand order checked by reading, not run on hardware
//
#include <hip/hip_runtime.h>
#include <hip/hip_bf16.h>

// CRF log-likelihood on MI455X (gfx1250).
// Forward recursion: new_alpha = log(exp(alpha-m) @ exp(T)) + m + emit, with the
// 128x128 logsumexp contraction on v_wmma_f32_16x16x32_f16.
// 32 workgroups x 2 waves (wave32). Each workgroup owns 16 batch rows; each wave
// owns 64 of the 128 output tags and keeps its 16 exp(T) B-fragments resident in
// VGPRs (128 VGPRs) for all 511 steps. alpha lives in LDS; emissions stream from
// HBM with global_prefetch_b8 one step ahead. Row-max uses an 8-way tree to cut
// the serial v_max chain; the row mask is bit-tested from one 8-byte register
// load instead of per-element LDS reads.

#define LSEQ 512
#define BSZ  512
#define TTAG 128

typedef __attribute__((ext_vector_type(16))) _Float16 v16h;
typedef __attribute__((ext_vector_type(8)))  float    v8f;
typedef __attribute__((ext_vector_type(4)))  float    v4f;

__global__ void crf_init_out(float* out) { out[0] = 0.0f; }

// ---------------------------------------------------------------------------
// Numerator: one thread per batch column, walk the tag path.
// ---------------------------------------------------------------------------
__global__ __launch_bounds__(256)
void crf_numerator(const float* __restrict__ emis,
                   const int* __restrict__ tags,
                   const unsigned char* __restrict__ mask,
                   const float* __restrict__ start_t,
                   const float* __restrict__ end_t,
                   const float* __restrict__ trans,
                   float* __restrict__ out) {
    int b = blockIdx.x * blockDim.x + threadIdx.x;
    if (b >= BSZ) return;
    float acc = start_t[tags[b]];
    int prev = 0, cnt = 0;
    for (int l = 0; l < LSEQ; ++l) {
        int tg = tags[l * BSZ + b];
        float sc = emis[((size_t)l * BSZ + b) * TTAG + tg];
        if (l > 0) sc += trans[prev * TTAG + tg];
        if (mask[l * BSZ + b]) { acc += sc; ++cnt; }
        prev = tg;
    }
    int lastrow = cnt > 0 ? cnt - 1 : 0;
    acc += end_t[tags[lastrow * BSZ + b]];
    atomicAdd(out, -acc * (1.0f / (float)BSZ));
}

// ---------------------------------------------------------------------------
// Forward algorithm: 2 waves per workgroup, 16 batch rows per workgroup.
// Wave w handles output tag columns [64w, 64w+64).
// ---------------------------------------------------------------------------
__global__ __launch_bounds__(64)
void crf_forward(const float* __restrict__ emis,
                 const unsigned char* __restrict__ mask,
                 const float* __restrict__ start_t,
                 const float* __restrict__ end_t,
                 const float* __restrict__ trans,
                 float* __restrict__ out) {
    __shared__ float alpha_s[16 * TTAG];   // 8 KB, shared by both waves
    __shared__ float mrow_s[16];

    const int tid  = threadIdx.x;
    const int wv   = tid >> 5;             // wave id: 0 or 1
    const int lane = tid & 31;
    const int g    = lane >> 4;            // lane half (K / row-block select)
    const int ln   = lane & 15;
    const int B0   = blockIdx.x * 16;

    // --- B operand: exp(transitions), resident in VGPRs for the whole loop --
    // bfr[nt_l*4 + kt] covers cols [16*(4w+nt_l), +16), K slice [32*kt, +32),
    // in the 16-bit B-operand lane layout (ISA 7.12.2).
    v16h bfr[16];
#pragma unroll
    for (int nt_l = 0; nt_l < 4; ++nt_l) {
        const int n = (4 * wv + nt_l) * 16 + ln;
#pragma unroll
        for (int kt = 0; kt < 4; ++kt) {
#pragma unroll
            for (int h = 0; h < 16; ++h) {
                const int k = kt * 32 + h + 8 * g + ((h >= 8) ? 8 : 0);
                bfr[nt_l * 4 + kt][h] = (_Float16)__expf(trans[k * TTAG + n]);
            }
        }
    }

    // --- alpha0 = start_transitions + emissions[0] -------------------------
    for (int idx = tid; idx < 16 * TTAG; idx += 64) {
        const int r = idx >> 7, t = idx & 127;
        alpha_s[idx] = start_t[t] + emis[((size_t)0 * BSZ + B0 + r) * TTAG + t];
    }
    __syncthreads();

    for (int i = 1; i < LSEQ; ++i) {
        // per-row max over 128 tags: lane owns row ln, half g -> 64 entries.
        // 8-way accumulator tree keeps the v_max dependence chain short.
        const float* ar = &alpha_s[ln * TTAG + g * 64];
        float mx[8];
#pragma unroll
        for (int j = 0; j < 8; ++j) mx[j] = ar[j];
#pragma unroll
        for (int t = 8; t < 64; t += 8)
#pragma unroll
            for (int j = 0; j < 8; ++j) mx[j] = fmaxf(mx[j], ar[t + j]);
        float m = fmaxf(fmaxf(fmaxf(mx[0], mx[1]), fmaxf(mx[2], mx[3])),
                        fmaxf(fmaxf(mx[4], mx[5]), fmaxf(mx[6], mx[7])));
        m = fmaxf(m, __shfl_xor(m, 16, 32));

        // A fragments: exp(alpha - m) for row ln, all K (reads alpha BEFORE
        // the mid-iteration barrier; writes happen after it)
        v16h afr[4];
#pragma unroll
        for (int kt = 0; kt < 4; ++kt) {
#pragma unroll
            for (int h = 0; h < 16; ++h) {
                const int k = kt * 32 + h + 8 * g + ((h >= 8) ? 8 : 0);
                afr[kt][h] = (_Float16)__expf(alpha_s[ln * TTAG + k] - m);
            }
        }

        // mask bytes for the 8 rows this lane will write (rows 8g..8g+7):
        // one aligned 8-byte load, bit-tested per row. No LDS involved.
        const unsigned long long mk8 =
            *(const unsigned long long*)&mask[(size_t)i * BSZ + B0 + 8 * g];

        if (wv == 0 && lane < 16) mrow_s[lane] = m;
        __syncthreads();   // alpha reads done; mrow published

        // acc[nt_l] = exp(alpha-m) @ exp(T)[:, cols of this wave]   (K = 128)
        v8f acc[4];
#pragma unroll
        for (int nt_l = 0; nt_l < 4; ++nt_l) {
            v8f c = {};
#pragma unroll
            for (int kt = 0; kt < 4; ++kt) {
                c = __builtin_amdgcn_wmma_f32_16x16x32_f16(
                        false, afr[kt], false, bfr[nt_l * 4 + kt],
                        (short)0, c, false, false);
            }
            acc[nt_l] = c;
        }

        const float* em = &emis[((size_t)i * BSZ + B0) * TTAG];
        if (i + 1 < LSEQ) {
            // pull next step's 8KB emission tile toward L2/L0 (64 threads x 128B)
            const float* nx = &emis[((size_t)(i + 1) * BSZ + B0) * TTAG];
            __builtin_prefetch(&nx[tid * 32], 0, 1);
        }

        // new_alpha = log(acc) + m_row + emit ; C/D layout: row = r + 8g
#pragma unroll
        for (int nt_l = 0; nt_l < 4; ++nt_l) {
            const int col = (4 * wv + nt_l) * 16 + ln;
#pragma unroll
            for (int r = 0; r < 8; ++r) {
                const int row2 = r + 8 * g;
                const float v = __log2f(acc[nt_l][r]) * 0.69314718055994531f
                              + mrow_s[row2] + em[row2 * TTAG + col];
                if ((mk8 >> (8 * r)) & 0xffull)
                    alpha_s[row2 * TTAG + col] = v;
            }
        }
        __syncthreads();   // writes done before next iteration's reads
    }

    // log_denominator: logsumexp(alpha + end_transitions) per row, summed.
    // Wave 0 only; lane owns row ln, half g.
    if (wv == 0) {
        float m2 = -3.402823466e38f;
        const float* ar = &alpha_s[ln * TTAG + g * 64];
        const float* er = &end_t[g * 64];
#pragma unroll
        for (int t = 0; t < 64; ++t) m2 = fmaxf(m2, ar[t] + er[t]);
        m2 = fmaxf(m2, __shfl_xor(m2, 16, 32));
        float s = 0.0f;
#pragma unroll
        for (int t = 0; t < 64; ++t) s += __expf(ar[t] + er[t] - m2);
        s += __shfl_xor(s, 16, 32);
        if (g == 0) {
            const float lse = m2 + __logf(s);
            atomicAdd(out, lse * (1.0f / (float)BSZ));
        }
    }
}

extern "C" void kernel_launch(void* const* d_in, const int* in_sizes, int n_in,
                              void* d_out, int out_size, void* d_ws, size_t ws_size,
                              hipStream_t stream) {
    const float*         emis    = (const float*)d_in[0];
    const int*           tags    = (const int*)d_in[1];
    const unsigned char* mask    = (const unsigned char*)d_in[2];
    const float*         start_t = (const float*)d_in[3];
    const float*         end_t   = (const float*)d_in[4];
    const float*         trans   = (const float*)d_in[5];
    float*               out     = (float*)d_out;

    crf_init_out<<<1, 1, 0, stream>>>(out);
    crf_numerator<<<(BSZ + 255) / 256, 256, 0, stream>>>(emis, tags, mask,
                                                         start_t, end_t, trans, out);
    crf_forward<<<BSZ / 16, 64, 0, stream>>>(emis, mask, start_t, end_t, trans, out);
}
